// YatPerformerAttention_55886114455727
// MI455X (gfx1250) — matrix operand order repaired and checked by reading, not compiled, not run on hardware
//
#include <hip/hip_runtime.h>

// YAT Performer attention for MI455X (gfx1250, wave32, WMMA bf16 path).
// All GEMMs run through v_wmma_f32_16x16x32_bf16; operands converted to bf16
// with f32 accumulation. "B" operands stored N-major (pre-transposed) so both
// fragments are contiguous 16B loads matching the ISA VGPR layouts.
// Big GEMMs (x@W projections, output proj) use 4x2 tiles/wave for higher
// arithmetic intensity; per-(b,h) GEMMs use 2x2.

#define EMBED 1024
#define HEADS 16
#define HDIM  64
#define FEATS 256
#define BATCH 4
#define SEQL  4096
#define ROWS  (BATCH*SEQL)   // 16384
#define BH    (BATCH*HEADS)  // 64
#define YAT_EPS 0.1f

typedef __attribute__((ext_vector_type(16))) __bf16 v16bf;
typedef __attribute__((ext_vector_type(8)))  __bf16 v8bf;
typedef __attribute__((ext_vector_type(8)))  float  v8f;

__device__ __forceinline__ v8bf ldg8(const __bf16* p) { return *(const v8bf*)p; }

__device__ __forceinline__ v16bf load_a_frag(const __bf16* row, int k0, int hl) {
  // A fragment (16x32): lanes0-15 hold K {0..7,16..23}; lanes16-31 {8..15,24..31}
  union { v16bf v; v8bf h[2]; } u;
  u.h[0] = ldg8(row + k0 + 8 * hl);
  u.h[1] = ldg8(row + k0 + 16 + 8 * hl);
  return u.v;
}

__device__ __forceinline__ v16bf load_b_frag(const __bf16* row, int k0, int hl) {
  // B fragment (32x16): lanes0-15 hold K {0..7,16..23}; lanes16-31 {8..15,24..31}
  union { v16bf v; v8bf h[2]; } u;
  u.h[0] = ldg8(row + k0 + 8 * hl);
  u.h[1] = ldg8(row + k0 + 16 + 8 * hl);
  return u.v;
}

// ---------------------------------------------------------------------------
// bf16 conversion / transpose-conversion kernels
// ---------------------------------------------------------------------------
__global__ void k_cvt(const float* __restrict__ in, __bf16* __restrict__ out, int n) {
  int i = blockIdx.x * blockDim.x + threadIdx.x;
  if (i < n) out[i] = (__bf16)in[i];
}

// W: (K=EMBED, N=EMBED) row-major -> Wt: (N, K) row-major, bf16
__global__ void k_cvt_tr(const float* __restrict__ W, __bf16* __restrict__ Wt) {
  int i = blockIdx.x * blockDim.x + threadIdx.x; // over EMBED*EMBED
  int k = i >> 10, n = i & (EMBED - 1);
  Wt[(size_t)n * EMBED + k] = (__bf16)W[i];
}

// ---------------------------------------------------------------------------
// 2x2 tile core (used by the small per-(b,h) GEMMs).
// A: (M,K) row-major bf16, lda=K.  Bt: (N,K) row-major bf16, ldb=K.
// ---------------------------------------------------------------------------
__device__ __forceinline__ void mma_2x2(const __bf16* __restrict__ A, int lda,
                                        const __bf16* __restrict__ Bt, int ldb,
                                        int m0, int n0, int K, v8f acc[2][2]) {
  const int lane = threadIdx.x & 31;
  const int hl   = lane >> 4;
  const int l16  = lane & 15;
  const __bf16* ar0 = A + (size_t)(m0 + l16) * lda;
  const __bf16* ar1 = A + (size_t)(m0 + 16 + l16) * lda;
  const __bf16* br0 = Bt + (size_t)(n0 + l16) * ldb;
  const __bf16* br1 = Bt + (size_t)(n0 + 16 + l16) * ldb;
  for (int k0 = 0; k0 < K; k0 += 32) {
    v16bf a0 = load_a_frag(ar0, k0, hl);
    v16bf a1 = load_a_frag(ar1, k0, hl);
    v16bf b0 = load_b_frag(br0, k0, hl);
    v16bf b1 = load_b_frag(br1, k0, hl);
    acc[0][0] = __builtin_amdgcn_wmma_f32_16x16x32_bf16(false, a0, false, b0, (short)0, acc[0][0], false, false);
    acc[0][1] = __builtin_amdgcn_wmma_f32_16x16x32_bf16(false, a0, false, b1, (short)0, acc[0][1], false, false);
    acc[1][0] = __builtin_amdgcn_wmma_f32_16x16x32_bf16(false, a1, false, b0, (short)0, acc[1][0], false, false);
    acc[1][1] = __builtin_amdgcn_wmma_f32_16x16x32_bf16(false, a1, false, b1, (short)0, acc[1][1], false, false);
  }
}

// ---------------------------------------------------------------------------
// Big GEMM: C(M,N) = A(M,K) x Bt(N,K)^T + bias, 4x2 tiles per wave.
// block = 128 threads (4 waves in a 2x2 grid), block covers 128M x 64N.
// mode 0: store f32 row-major. mode 1: store bf16 scattered as V^T (b,h,d,s).
// ---------------------------------------------------------------------------
__global__ void k_gemm(const __bf16* __restrict__ A, const __bf16* __restrict__ Bt,
                       const float* __restrict__ bias, float* __restrict__ C,
                       __bf16* __restrict__ Vt, int M, int N, int K, int mode) {
  const int wave = threadIdx.x >> 5;
  const int wr = wave >> 1, wc = wave & 1;
  const int lane = threadIdx.x & 31;
  const int hl = lane >> 4, l16 = lane & 15;
  const int mBase = blockIdx.y * 128 + wr * 64;  // 4 M-tiles of 16
  const int nBase = blockIdx.x * 64 + wc * 32;   // 2 N-tiles of 16
  v8f acc[4][2] = {};

  const __bf16* arow[4];
#pragma unroll
  for (int i = 0; i < 4; ++i) arow[i] = A + (size_t)(mBase + i * 16 + l16) * K;
  const __bf16* brow[2];
#pragma unroll
  for (int j = 0; j < 2; ++j) brow[j] = Bt + (size_t)(nBase + j * 16 + l16) * K;

  for (int k0 = 0; k0 < K; k0 += 32) {
    v16bf af[4], bf[2];
#pragma unroll
    for (int i = 0; i < 4; ++i) af[i] = load_a_frag(arow[i], k0, hl);
#pragma unroll
    for (int j = 0; j < 2; ++j) bf[j] = load_b_frag(brow[j], k0, hl);
#pragma unroll
    for (int i = 0; i < 4; ++i)
#pragma unroll
      for (int j = 0; j < 2; ++j)
        acc[i][j] = __builtin_amdgcn_wmma_f32_16x16x32_bf16(
            false, af[i], false, bf[j], (short)0, acc[i][j], false, false);
  }

  if (mode == 0) {
#pragma unroll
    for (int i = 0; i < 4; ++i)
#pragma unroll
      for (int j = 0; j < 2; ++j) {
        const int n  = nBase + j * 16 + l16;
        const int mb = mBase + i * 16 + 8 * hl;
        const float badd = bias[n];
#pragma unroll
        for (int r = 0; r < 8; ++r)
          C[(size_t)(mb + r) * N + n] = acc[i][j][r] + badd;
      }
  } else { // V^T: (b,h,d,s) bf16
#pragma unroll
    for (int i = 0; i < 4; ++i)
#pragma unroll
      for (int j = 0; j < 2; ++j) {
        const int n  = nBase + j * 16 + l16;
        const int mb = mBase + i * 16 + 8 * hl;
        const float badd = bias[n];
        const int h = n >> 6, d = n & (HDIM - 1);
#pragma unroll
        for (int r = 0; r < 8; ++r) {
          const int m = mb + r;
          const int b = m >> 12, s = m & (SEQL - 1);
          Vt[(((size_t)(b * HEADS + h)) * HDIM + d) * SEQL + s] = (__bf16)(acc[i][j][r] + badd);
        }
      }
  }
}

// ---------------------------------------------------------------------------
// Per-head L2 normalize: X (ROWS, EMBED) f32 -> Xn (b,h,s,d) bf16. 1 wave/(row,h)
// ---------------------------------------------------------------------------
__global__ void k_norm_heads(const float* __restrict__ X, __bf16* __restrict__ Xn) {
  const int gw = (blockIdx.x * blockDim.x + threadIdx.x) >> 5;
  const int lane = threadIdx.x & 31;
  if (gw >= ROWS * HEADS) return;
  const int row = gw / HEADS, h = gw % HEADS;
  const float* p = X + (size_t)row * EMBED + h * HDIM;
  const float x0 = p[lane], x1 = p[lane + 32];
  float ss = x0 * x0 + x1 * x1;
#pragma unroll
  for (int off = 16; off; off >>= 1) ss += __shfl_xor(ss, off);
  const float inv = 1.0f / (sqrtf(ss) + 1e-8f);
  const int b = row >> 12, s = row & (SEQL - 1);
  __bf16* q = Xn + (((size_t)(b * HEADS + h)) * SEQL + s) * HDIM;
  q[lane]      = (__bf16)(x0 * inv);
  q[lane + 32] = (__bf16)(x1 * inv);
}

// ---------------------------------------------------------------------------
// Feature GEMM per (b,h): (SEQ x HDIM) x (FEATS x HDIM)^T, relu + eps.
// trans=0: store row-major (s,f) bf16.  trans=1: store (f,s) bf16.
// ---------------------------------------------------------------------------
__global__ void k_feat(const __bf16* __restrict__ Xn, const __bf16* __restrict__ P,
                       __bf16* __restrict__ out, int trans) {
  const int bh = blockIdx.z, h = bh % HEADS;
  const __bf16* A  = Xn + (size_t)bh * SEQL * HDIM;
  const __bf16* Bt = P + (size_t)h * FEATS * HDIM;
  const int wave = threadIdx.x >> 5;
  const int wr = wave >> 1, wc = wave & 1;
  const int lane = threadIdx.x & 31;
  const int hl = lane >> 4, l16 = lane & 15;
  const int mBase = blockIdx.y * 64 + wr * 32;   // over SEQ
  const int nBase = blockIdx.x * 64 + wc * 32;   // over FEATS
  v8f acc[2][2] = {};
  mma_2x2(A, HDIM, Bt, HDIM, mBase, nBase, HDIM, acc);
#pragma unroll
  for (int i = 0; i < 2; ++i)
#pragma unroll
    for (int j = 0; j < 2; ++j) {
      const int n  = nBase + j * 16 + l16;
      const int mb = mBase + i * 16 + 8 * hl;
#pragma unroll
      for (int r = 0; r < 8; ++r) {
        const int m = mb + r;
        const float v = fmaxf(acc[i][j][r], 0.0f) + YAT_EPS;
        if (!trans) out[(size_t)bh * SEQL * FEATS + (size_t)m * FEATS + n] = (__bf16)v;
        else        out[(size_t)bh * FEATS * SEQL + (size_t)n * SEQL + m] = (__bf16)v;
      }
    }
}

// ---------------------------------------------------------------------------
// KV GEMM per (b,h): (FEATS x SEQ) x (HDIM x SEQ)^T -> store KV^T (d,f) bf16
// ---------------------------------------------------------------------------
__global__ void k_kv(const __bf16* __restrict__ Kpt, const __bf16* __restrict__ Vt,
                     __bf16* __restrict__ KVt) {
  const int bh = blockIdx.z;
  const __bf16* A  = Kpt + (size_t)bh * FEATS * SEQL; // (f,s)
  const __bf16* Bt = Vt  + (size_t)bh * HDIM * SEQL;  // (d,s)
  const int wave = threadIdx.x >> 5;
  const int wr = wave >> 1, wc = wave & 1;
  const int lane = threadIdx.x & 31;
  const int hl = lane >> 4, l16 = lane & 15;
  const int mBase = blockIdx.y * 64 + wr * 32;  // over FEATS
  const int nBase = blockIdx.x * 64 + wc * 32;  // over HDIM
  v8f acc[2][2] = {};
  mma_2x2(A, SEQL, Bt, SEQL, mBase, nBase, SEQL, acc);
#pragma unroll
  for (int i = 0; i < 2; ++i)
#pragma unroll
    for (int j = 0; j < 2; ++j) {
      const int n  = nBase + j * 16 + l16;
      const int mb = mBase + i * 16 + 8 * hl;
#pragma unroll
      for (int r = 0; r < 8; ++r)
        KVt[(size_t)bh * HDIM * FEATS + (size_t)n * FEATS + (mb + r)] = (__bf16)acc[i][j][r];
    }
}

// ---------------------------------------------------------------------------
// QKV GEMM per (b,h): (SEQ x FEATS) x (HDIM x FEATS)^T -> QKV (s,d) f32
// ---------------------------------------------------------------------------
__global__ void k_qkv(const __bf16* __restrict__ Qp, const __bf16* __restrict__ KVt,
                      float* __restrict__ QKV) {
  const int bh = blockIdx.z;
  const __bf16* A  = Qp  + (size_t)bh * SEQL * FEATS;
  const __bf16* Bt = KVt + (size_t)bh * HDIM * FEATS;
  const int wave = threadIdx.x >> 5;
  const int wr = wave >> 1, wc = wave & 1;
  const int lane = threadIdx.x & 31;
  const int hl = lane >> 4, l16 = lane & 15;
  const int mBase = blockIdx.y * 64 + wr * 32;  // over SEQ
  const int nBase = blockIdx.x * 64 + wc * 32;  // over HDIM
  v8f acc[2][2] = {};
  mma_2x2(A, FEATS, Bt, FEATS, mBase, nBase, FEATS, acc);
#pragma unroll
  for (int i = 0; i < 2; ++i)
#pragma unroll
    for (int j = 0; j < 2; ++j) {
      const int n  = nBase + j * 16 + l16;
      const int mb = mBase + i * 16 + 8 * hl;
#pragma unroll
      for (int r = 0; r < 8; ++r)
        QKV[(size_t)bh * SEQL * HDIM + (size_t)(mb + r) * HDIM + n] = acc[i][j][r];
    }
}

// ---------------------------------------------------------------------------
// K_sum[b,h,f] = sum_s Kpt[bh,f,s]. One wave per row.
// ---------------------------------------------------------------------------
__global__ void k_ksum(const __bf16* __restrict__ Kpt, float* __restrict__ Ksum) {
  const int gw = (blockIdx.x * blockDim.x + threadIdx.x) >> 5;
  const int lane = threadIdx.x & 31;
  if (gw >= BH * FEATS) return;
  const __bf16* p = Kpt + (size_t)gw * SEQL;
  float s = 0.0f;
  for (int i = lane; i < SEQL; i += 32) s += (float)p[i];
#pragma unroll
  for (int off = 16; off; off >>= 1) s += __shfl_xor(s, off);
  if (lane == 0) Ksum[gw] = s;
}

// ---------------------------------------------------------------------------
// Normalizer + pack: Ob[(b,s),(h,d)] bf16 = QKV / max(Qp . Ksum, 1e-6)
// One wave per (bh, s).
// ---------------------------------------------------------------------------
__global__ void k_finalize(const __bf16* __restrict__ Qp, const float* __restrict__ Ksum,
                           const float* __restrict__ QKV, __bf16* __restrict__ Ob) {
  const int gw = (blockIdx.x * blockDim.x + threadIdx.x) >> 5;
  const int lane = threadIdx.x & 31;
  if (gw >= BH * SEQL) return;
  const int bh = gw >> 12, s = gw & (SEQL - 1);
  const __bf16* qp = Qp + ((size_t)bh * SEQL + s) * FEATS;
  const float* ks = Ksum + (size_t)bh * FEATS;
  float acc = 0.0f;
#pragma unroll
  for (int i = 0; i < FEATS / 32; ++i) {
    const int f = lane + i * 32;
    acc += (float)qp[f] * ks[f];
  }
#pragma unroll
  for (int off = 16; off; off >>= 1) acc += __shfl_xor(acc, off);
  const float inv = 1.0f / fmaxf(acc, 1e-6f);
  const int b = bh / HEADS, h = bh % HEADS;
  const float* qkv = QKV + ((size_t)bh * SEQL + s) * HDIM;
  __bf16* o = Ob + ((size_t)(b * SEQL + s)) * EMBED + h * HDIM;
  o[lane]      = (__bf16)(qkv[lane] * inv);
  o[lane + 32] = (__bf16)(qkv[lane + 32] * inv);
}

// ---------------------------------------------------------------------------
static inline size_t al256(size_t x) { return (x + 255) & ~(size_t)255; }

extern "C" void kernel_launch(void* const* d_in, const int* in_sizes, int n_in,
                              void* d_out, int out_size, void* d_ws, size_t ws_size,
                              hipStream_t stream) {
  const float* x    = (const float*)d_in[0];
  const float* Wq   = (const float*)d_in[1];
  const float* bq   = (const float*)d_in[2];
  const float* Wk   = (const float*)d_in[3];
  const float* bk   = (const float*)d_in[4];
  const float* Wv   = (const float*)d_in[5];
  const float* bv   = (const float*)d_in[6];
  const float* Wo   = (const float*)d_in[7];
  const float* bo   = (const float*)d_in[8];
  const float* proj = (const float*)d_in[9];

  char* w = (char*)d_ws;
  auto carve = [&](size_t bytes) { void* p = (void*)w; w += al256(bytes); return p; };

  __bf16* xb   = (__bf16*)carve((size_t)ROWS * EMBED * 2);
  __bf16* Wqb  = (__bf16*)carve((size_t)EMBED * EMBED * 2);
  __bf16* Wkb  = (__bf16*)carve((size_t)EMBED * EMBED * 2);
  __bf16* Wvb  = (__bf16*)carve((size_t)EMBED * EMBED * 2);
  __bf16* Wob  = (__bf16*)carve((size_t)EMBED * EMBED * 2);
  __bf16* Pb   = (__bf16*)carve((size_t)HEADS * FEATS * HDIM * 2);
  float*  Qf   = (float*) carve((size_t)ROWS * EMBED * 4);
  float*  Kf   = (float*) carve((size_t)ROWS * EMBED * 4);
  __bf16* Vt   = (__bf16*)carve((size_t)BH * HDIM * SEQL * 2);
  __bf16* Qn   = (__bf16*)carve((size_t)BH * SEQL * HDIM * 2);
  __bf16* Kn   = (__bf16*)carve((size_t)BH * SEQL * HDIM * 2);
  __bf16* Qp   = (__bf16*)carve((size_t)BH * SEQL * FEATS * 2);
  __bf16* Kpt  = (__bf16*)carve((size_t)BH * FEATS * SEQL * 2);
  __bf16* KVt  = (__bf16*)carve((size_t)BH * HDIM * FEATS * 2);
  float*  Ksum = (float*) carve((size_t)BH * FEATS * 4);
  // Aliases (lifetimes don't overlap on the serialized stream):
  float*  QKV  = Qf;            // QKV f32 reuses Q f32 buffer
  __bf16* Ob   = (__bf16*)Kf;   // packed bf16 output reuses K f32 buffer
  (void)ws_size; (void)n_in; (void)in_sizes; (void)out_size;

  // 1. bf16 conversions
  k_cvt<<<(ROWS * EMBED) / 256, 256, 0, stream>>>(x, xb, ROWS * EMBED);
  k_cvt_tr<<<(EMBED * EMBED) / 256, 256, 0, stream>>>(Wq, Wqb);
  k_cvt_tr<<<(EMBED * EMBED) / 256, 256, 0, stream>>>(Wk, Wkb);
  k_cvt_tr<<<(EMBED * EMBED) / 256, 256, 0, stream>>>(Wv, Wvb);
  k_cvt_tr<<<(EMBED * EMBED) / 256, 256, 0, stream>>>(Wo, Wob);
  k_cvt<<<(HEADS * FEATS * HDIM) / 256, 256, 0, stream>>>(proj, Pb, HEADS * FEATS * HDIM);

  // 2. Q/K/V projections (WMMA, 4x2 tiles/wave, 128x64 per block)
  dim3 gBig(EMBED / 64, ROWS / 128);
  k_gemm<<<gBig, 128, 0, stream>>>(xb, Wqb, bq, Qf, nullptr, ROWS, EMBED, EMBED, 0);
  k_gemm<<<gBig, 128, 0, stream>>>(xb, Wkb, bk, Kf, nullptr, ROWS, EMBED, EMBED, 0);
  k_gemm<<<gBig, 128, 0, stream>>>(xb, Wvb, bv, nullptr, Vt, ROWS, EMBED, EMBED, 1);

  // 3. per-head L2 normalize -> bf16
  const int nrmBlocks = (ROWS * HEADS) / 8; // 8 waves per 256-thread block
  k_norm_heads<<<nrmBlocks, 256, 0, stream>>>(Qf, Qn);
  k_norm_heads<<<nrmBlocks, 256, 0, stream>>>(Kf, Kn);

  // 4. feature GEMMs (WMMA) with relu+eps epilogue
  dim3 gFeat(FEATS / 64, SEQL / 64, BH);
  k_feat<<<gFeat, 128, 0, stream>>>(Qn, Pb, Qp, 0);   // (s,f)
  k_feat<<<gFeat, 128, 0, stream>>>(Kn, Pb, Kpt, 1);  // (f,s)

  // 5. K_sum reduction
  k_ksum<<<(BH * FEATS) / 8, 256, 0, stream>>>(Kpt, Ksum);

  // 6. KV = Kp^T V (WMMA), store KV^T
  dim3 gKV(HDIM / 64, FEATS / 64, BH);
  k_kv<<<gKV, 128, 0, stream>>>(Kpt, Vt, KVt);

  // 7. QKV = Qp KV (WMMA)
  dim3 gQKV(HDIM / 64, SEQL / 64, BH);
  k_qkv<<<gQKV, 128, 0, stream>>>(Qp, KVt, QKV);

  // 8. normalizer + divide + pack to (b,s,embed) bf16
  k_finalize<<<(BH * SEQL) / 8, 256, 0, stream>>>(Qp, Ksum, QKV, Ob);

  // 9. output projection (WMMA) -> d_out f32
  k_gemm<<<gBig, 128, 0, stream>>>(Ob, Wob, bo, (float*)d_out, nullptr, ROWS, EMBED, EMBED, 0);
}
